// EncoderRNN_41205916237884
// MI455X (gfx1250) — matrix operand order repaired and checked by reading, not compiled, not run on hardware
//
#include <hip/hip_runtime.h>
#include <hip/hip_bf16.h>

// ---------------------------------------------------------------------------
// EncoderRNN for MI455X (gfx1250, wave32, WMMA bf16 + async-to-LDS path)
// B=64, S=512, V=32000, E=256, H=512
// ---------------------------------------------------------------------------

#define BATCH 64
#define SEQ   512
#define EDIM  256
#define HDIM  512
#define TOKENS (BATCH * SEQ)          // 32768
#define YSZ   ((size_t)TOKENS * HDIM) // 16,777,216 floats

typedef __attribute__((ext_vector_type(16))) __bf16 v16bf;
typedef __attribute__((ext_vector_type(8)))  __bf16 v8bf;
typedef __attribute__((ext_vector_type(8)))  float  v8f;

#define WMMA_BF16(a, b, c) \
    __builtin_amdgcn_wmma_f32_16x16x32_bf16(false, (a), false, (b), (short)0, (c), false, false)

// Async copy 16B global -> LDS, tracked by ASYNCcnt (gfx1250 GLOBAL_LOAD_ASYNC_TO_LDS_B128).
// ldsoff = LDS byte address (low 32 bits of the generic shared pointer, per the
// gfx1250 aperture layout), gaddr = 64-bit global address.
#define ASYNC_G2L_B128(ldsoff, gaddr)                                           \
    asm volatile("global_load_async_to_lds_b128 %0, %1, off"                    \
                 :: "v"((unsigned)(ldsoff)), "v"((unsigned long long)(gaddr))   \
                 : "memory")

// ---------------------------------------------------------------------------
// Kernel 1: embedding gather + f32 -> bf16 cast.  One thread = 4 elements.
// ---------------------------------------------------------------------------
__global__ __launch_bounds__(256)
void gather_cast_kernel(const int* __restrict__ src,
                        const float* __restrict__ emb,
                        __bf16* __restrict__ x) {
    int tid   = blockIdx.x * blockDim.x + threadIdx.x;  // 0 .. TOKENS*64-1
    int token = tid >> 6;                               // 64 threads per token (E/4)
    int e4    = (tid & 63) << 2;
    const float4 v = *(const float4*)(emb + (size_t)src[token] * EDIM + e4);
    __bf16* o = x + (size_t)token * EDIM + e4;
    o[0] = (__bf16)v.x; o[1] = (__bf16)v.y; o[2] = (__bf16)v.z; o[3] = (__bf16)v.w;
}

// ---------------------------------------------------------------------------
// Kernel 2: f32 -> bf16 weight cast (vectorized by 4)
// ---------------------------------------------------------------------------
__global__ __launch_bounds__(256)
void cast_bf16_kernel(const float* __restrict__ in, __bf16* __restrict__ out, int n4) {
    int i = blockIdx.x * blockDim.x + threadIdx.x;
    if (i < n4) {
        float4 v = ((const float4*)in)[i];
        __bf16* o = out + 4 * (size_t)i;
        o[0] = (__bf16)v.x; o[1] = (__bf16)v.y; o[2] = (__bf16)v.z; o[3] = (__bf16)v.w;
    }
}

// ---------------------------------------------------------------------------
// Kernel 3: C[M,N] = A[M,K](bf16) * W[N,K]^T(bf16) + bias_a[N] + bias_b[N]
// 8 waves / block; each wave computes a 16(M) x 64(N) tile: one A fetch feeds
// FOUR v_wmma_f32_16x16x32_bf16 per K step (2x the FLOP/byte of a 16x16 tile).
// ---------------------------------------------------------------------------
__global__ __launch_bounds__(256)
void wmma_gemm_bf16_kernel(const __bf16* __restrict__ A,
                           const __bf16* __restrict__ W,
                           const float* __restrict__ bias_a,
                           const float* __restrict__ bias_b,
                           float* __restrict__ C,
                           int N, int K) {
    const int lane = threadIdx.x & 31;
    const int wave = threadIdx.x >> 5;          // 0..7
    const int half = lane >> 4;                 // 0 or 1 (lane group)
    const int l15  = lane & 15;

    const int ntn  = N >> 6;                    // 64-wide N tiles
    const int tile = blockIdx.x * 8 + wave;
    const int m0   = (tile / ntn) << 4;
    const int n0   = (tile % ntn) << 6;

    // A: lane holds row m0+l15; 16 contiguous K at half*16 (per 32-K step)
    const __bf16* arow = A + (size_t)(m0 + l15) * K + half * 16;
    // B: lane holds column n of W^T == row n of W; 16 contiguous K per fetch
    const __bf16* w0 = W + (size_t)(n0 +  0 + l15) * K + half * 16;
    const __bf16* w1 = W + (size_t)(n0 + 16 + l15) * K + half * 16;
    const __bf16* w2 = W + (size_t)(n0 + 32 + l15) * K + half * 16;
    const __bf16* w3 = W + (size_t)(n0 + 48 + l15) * K + half * 16;

    v8f acc0 = {}, acc1 = {}, acc2 = {}, acc3 = {};
    for (int k = 0; k < K; k += 32) {
        __builtin_prefetch(arow + k + 256, 0, 1);   // global_prefetch_b8 ahead of K loop
        v16bf a;
        ((v8bf*)&a)[0] = *(const v8bf*)(arow + k);
        ((v8bf*)&a)[1] = *(const v8bf*)(arow + k + 8);
        acc0 = WMMA_BF16(a, *(const v16bf*)(w0 + k), acc0);
        acc1 = WMMA_BF16(a, *(const v16bf*)(w1 + k), acc1);
        acc2 = WMMA_BF16(a, *(const v16bf*)(w2 + k), acc2);
        acc3 = WMMA_BF16(a, *(const v16bf*)(w3 + k), acc3);
    }

    v8f accs[4] = {acc0, acc1, acc2, acc3};
#pragma unroll
    for (int j = 0; j < 4; ++j) {
        const int nc = n0 + j * 16 + l15;
        const float bi = bias_a[nc] + bias_b[nc];
        float* crow = C + (size_t)(m0 + half * 8) * N + nc;
#pragma unroll
        for (int r = 0; r < 8; ++r)
            crow[(size_t)r * N] = accs[j][r] + bi;
    }
}

// ---------------------------------------------------------------------------
// Kernel 4: persistent recurrence  h_t = tanh(pre_t + h_{t-1} @ Whh^T)
// grid = 4 blocks (batch tiles of 16 rows), block = 1024 threads = 32 waves.
// Wave w owns output columns [16w,16w+16); its 512x16 Whh^T slice is register
// resident (16 x v16bf = 128 VGPRs).  Hidden state (16x512 bf16) lives in LDS.
// The next step's 16x512 f32 pre-tile is double-buffered in LDS via
// global_load_async_to_lds_b128 (ASYNCcnt), overlapped with the WMMA chain so
// no global-memory latency sits on the sequential critical path.
// LDS: 16KB h + 2*32KB pre tiles = 80KB (of 320KB/WGP).
// ---------------------------------------------------------------------------
__global__ __launch_bounds__(1024)
void rnn_recur_kernel(const float* __restrict__ pre,     // [B,S,H]
                      const __bf16* __restrict__ Whh,    // [H,H] row-major (N,K)
                      float* __restrict__ yf,            // final layer out [B,S,H] or null
                      __bf16* __restrict__ yb,           // intermediate out (bf16) or null
                      float* __restrict__ hlast,         // [B,H] slice for this layer
                      int final_layer) {
    __shared__ __bf16 hbuf[16][HDIM];                    // 16 KB
    __shared__ float  ptile[2][16][HDIM];                // 64 KB, double-buffered pre tile

    const int lane = threadIdx.x & 31;
    const int wave = threadIdx.x >> 5;                   // 0..31
    const int n0   = wave << 4;
    const int half = lane >> 4;
    const int l15  = lane & 15;
    const int b0   = blockIdx.x << 4;                    // batch tile base
    const int mrow = half * 8;

    // async-prefetch geometry: thread i copies 32B of row (i>>6), cols (i&63)*8..+8
    const int pm = threadIdx.x >> 6;                     // 0..15
    const int pc = (threadIdx.x & 63) << 3;              // 0,8,..,504
    const float*   gbase = pre + ((size_t)(b0 + pm) * SEQ) * HDIM + pc;
    const unsigned l0    = (unsigned)(uintptr_t)&ptile[0][pm][pc];
    const unsigned l1    = (unsigned)(uintptr_t)&ptile[1][pm][pc];

    // h0 = 0
    for (int i = threadIdx.x; i < 16 * HDIM / 2; i += 1024)
        ((unsigned int*)hbuf)[i] = 0u;

    // kick off tile t=0 into buffer 0
    ASYNC_G2L_B128(l0,      gbase);
    ASYNC_G2L_B128(l0 + 16, gbase + 4);

    // Preload this wave's Whh^T slice into registers (column n0+l15, all K)
    v16bf bc[16];
    {
        const __bf16* wr = Whh + (size_t)(n0 + l15) * HDIM + half * 16;
#pragma unroll
        for (int kc = 0; kc < 16; ++kc)
            bc[kc] = *(const v16bf*)(wr + kc * 32);
    }
    __syncthreads();

    const __bf16* hrow = &hbuf[l15][0] + half * 16;      // A operand source row

    for (int t = 0; t < SEQ; ++t) {
        const int cur = t & 1;

        // 1) prefetch next step's pre tile into the other buffer (uniform branch)
        if (t < SEQ - 1) {
            const unsigned ln = (cur ? l0 : l1);
            const float*   gn = gbase + (size_t)(t + 1) * HDIM;
            ASYNC_G2L_B128(ln,      gn);
            ASYNC_G2L_B128(ln + 16, gn + 4);
        }

        // 2) h_{t-1} @ Whh^T  (reads hbuf; Whh slice in registers)
        v8f acc = {};
#pragma unroll
        for (int kc = 0; kc < 16; ++kc) {
            v16bf a;
            ((v8bf*)&a)[0] = *(const v8bf*)(hrow + kc * 32);        // ds_load_b128
            ((v8bf*)&a)[1] = *(const v8bf*)(hrow + kc * 32 + 8);    // ds_load_b128
            acc = WMMA_BF16(a, bc[kc], acc);
        }

        // 3) retire this step's async tile: each wave issued 2 ops/step, in-order
        //    completion => waiting to <=2 retires tile t (last step: wait to 0)
        if (t < SEQ - 1) asm volatile("s_wait_asynccnt 0x2" ::: "memory");
        else             asm volatile("s_wait_asynccnt 0x0" ::: "memory");
        __syncthreads();   // all waves: tile t visible, all old-h reads done

        // 4) add input projection (from LDS), tanh
        float hv[8];
#pragma unroll
        for (int r = 0; r < 8; ++r)
            hv[r] = tanhf(acc[r] + ptile[cur][mrow + r][n0 + l15]);

        // 5) write new h + outputs
#pragma unroll
        for (int r = 0; r < 8; ++r) {
            const int m = mrow + r;
            hbuf[m][n0 + l15] = (__bf16)hv[r];
            const size_t oidx = ((size_t)(b0 + m) * SEQ + (size_t)t) * HDIM + n0 + l15;
            if (final_layer) yf[oidx] = hv[r];
            else             yb[oidx] = (__bf16)hv[r];
            if (t == SEQ - 1)
                hlast[(size_t)(b0 + m) * HDIM + n0 + l15] = hv[r];
        }
        __syncthreads();   // new h visible; ptile[cur] fully consumed before re-prefetch
    }
}

// ---------------------------------------------------------------------------
// Host launcher
// ---------------------------------------------------------------------------
extern "C" void kernel_launch(void* const* d_in, const int* in_sizes, int n_in,
                              void* d_out, int out_size, void* d_ws, size_t ws_size,
                              hipStream_t stream) {
    const int*   src   = (const int*)  d_in[0];
    const float* emb   = (const float*)d_in[1];
    const float* W_ih0 = (const float*)d_in[2];
    const float* W_hh0 = (const float*)d_in[3];
    const float* b_ih0 = (const float*)d_in[4];
    const float* b_hh0 = (const float*)d_in[5];
    const float* W_ih1 = (const float*)d_in[6];
    const float* W_hh1 = (const float*)d_in[7];
    const float* b_ih1 = (const float*)d_in[8];
    const float* b_hh1 = (const float*)d_in[9];
    float* out = (float*)d_out;

    // workspace carve-up (all offsets 64B-aligned)
    char* ws = (char*)d_ws;
    __bf16* xb     = (__bf16*)(ws);                    // 32768*256*2   = 16,777,216
    __bf16* wih0b  = (__bf16*)(ws + 16777216);         // 512*256*2     =    262,144
    __bf16* whh0b  = (__bf16*)(ws + 17039360);         // 512*512*2     =    524,288
    __bf16* wih1b  = (__bf16*)(ws + 17563648);         // 512*512*2     =    524,288
    __bf16* whh1b  = (__bf16*)(ws + 18087936);         // 512*512*2     =    524,288
    float*  prebuf = (float*) (ws + 18612224);         // 32768*512*4   = 67,108,864 (reused L0/L1)
    __bf16* y0b    = (__bf16*)(ws + 85721088);         // 32768*512*2   = 33,554,432

    // 1) embedding gather + cast  (TOKENS*E/4 threads)
    gather_cast_kernel<<<TOKENS * (EDIM / 4) / 256, 256, 0, stream>>>(src, emb, xb);

    // 2) weight casts
    cast_bf16_kernel<<<(HDIM * EDIM / 4 + 255) / 256, 256, 0, stream>>>(W_ih0, wih0b, HDIM * EDIM / 4);
    cast_bf16_kernel<<<(HDIM * HDIM / 4 + 255) / 256, 256, 0, stream>>>(W_hh0, whh0b, HDIM * HDIM / 4);
    cast_bf16_kernel<<<(HDIM * HDIM / 4 + 255) / 256, 256, 0, stream>>>(W_ih1, wih1b, HDIM * HDIM / 4);
    cast_bf16_kernel<<<(HDIM * HDIM / 4 + 255) / 256, 256, 0, stream>>>(W_hh1, whh1b, HDIM * HDIM / 4);

    // 3) layer-0 input projection: pre0 = x @ W_ih0^T + b_ih0 + b_hh0
    //    wave-tiles = (32768/16)*(512/64) = 16384; 8 waves/block -> 2048 blocks
    wmma_gemm_bf16_kernel<<<(TOKENS / 16) * (HDIM / 64) / 8, 256, 0, stream>>>(
        xb, wih0b, b_ih0, b_hh0, prebuf, HDIM, EDIM);

    // 4) layer-0 recurrence -> y0 (bf16), h_last0
    rnn_recur_kernel<<<BATCH / 16, 1024, 0, stream>>>(
        prebuf, whh0b, nullptr, y0b, out + YSZ, 0);

    // 5) layer-1 input projection: pre1 = y0 @ W_ih1^T + b_ih1 + b_hh1
    wmma_gemm_bf16_kernel<<<(TOKENS / 16) * (HDIM / 64) / 8, 256, 0, stream>>>(
        y0b, wih1b, b_ih1, b_hh1, prebuf, HDIM, HDIM);

    // 6) layer-1 recurrence -> y1 (f32, final output), h_last1
    rnn_recur_kernel<<<BATCH / 16, 1024, 0, stream>>>(
        prebuf, whh1b, out, nullptr, out + YSZ + (size_t)BATCH * HDIM, 1);
}
